// MemoryAccessPredictor_18975165513983
// MI455X (gfx1250) — matrix operand order repaired and checked by reading, not compiled, not run on hardware
//
#include <hip/hip_runtime.h>
#include <stdint.h>

#define BATCH 8192
#define TSEQ  32
#define FDIM  64
#define HDIM  128
#define GDIM  512   // 4*HDIM
#define BM    64    // batch rows per workgroup in LSTM kernels

typedef __bf16 v16bf __attribute__((ext_vector_type(16)));
typedef float  v8f   __attribute__((ext_vector_type(8)));

union FragU { uint32_t u[8]; v16bf v; };

__device__ __forceinline__ uint16_t f2bf(float x) {
  uint32_t u = __float_as_uint(x);
  u += 0x7FFFu + ((u >> 16) & 1u);   // round to nearest even
  return (uint16_t)(u >> 16);
}
__device__ __forceinline__ float sigm(float x) { return 1.0f / (1.0f + __expf(-x)); }

// ---------------------------------------------------------------------------
// A-fragment (16x32 bf16, M x K): lanes hold M=lane&15; lane group (lane>>4)
// selects K offset +8. VGPR v: K = (v>=4?16:0) + 2*(v&3) + 8*(lane>>4), pair (k,k+1).
// ---------------------------------------------------------------------------
__device__ __forceinline__ v16bf load_A_frag(const uint16_t* s, int ldK, int m, int kbase, int g2) {
  FragU f;
  const uint32_t* row = (const uint32_t*)(s + (size_t)m * ldK + kbase);
#pragma unroll
  for (int v = 0; v < 8; ++v) {
    int k = ((v >= 4) ? 16 : 0) + 2 * (v & 3) + 8 * g2;
    f.u[v] = row[k >> 1];
  }
  return f.v;
}

// B-fragment (32x16 bf16, K x N) pre-packed: lane l owns words [l*8, l*8+8)
// -> two ds_load_b128 per fragment.
__device__ __forceinline__ v16bf load_B_frag(const uint32_t* packed, int frag, int lane) {
  FragU f;
  const uint32_t* p = packed + (size_t)frag * 256 + lane * 8;
  *(uint4*)(&f.u[0]) = *(const uint4*)(p);
  *(uint4*)(&f.u[4]) = *(const uint4*)(p + 4);
  return f.v;
}

__device__ __forceinline__ v8f wmma_bf16(v16bf a, v16bf b, v8f c) {
  return __builtin_amdgcn_wmma_f32_16x16x32_bf16(false, a, false, b, (short)0, c, false, false);
}

// CDNA5 async global->LDS copy (ASYNCcnt-tracked DMA path). All 256 threads
// issue b128 transfers; bytes must be a multiple of 16*256 is NOT required --
// loop strides by 4KB (256 lanes * 16B).
__device__ __forceinline__ void async_copy_tile(const uint16_t* gsrc, uint16_t* lds_dst,
                                                int bytes, int tid) {
  const uint64_t base  = (uint64_t)(uintptr_t)gsrc;
  const uint32_t lbase = (uint32_t)(uintptr_t)lds_dst;
  for (int i = tid * 16; i < bytes; i += 256 * 16) {
    uint32_t la = lbase + (uint32_t)i;
    uint32_t go = (uint32_t)i;
    asm volatile("global_load_async_to_lds_b128 %0, %1, %2 offset:0"
                 :: "v"(la), "v"(go), "s"(base) : "memory");
  }
}
__device__ __forceinline__ void wait_async0() {
  asm volatile("s_wait_asynccnt 0x0" ::: "memory");
}

// ---------------------------------------------------------------------------
// Pack W (GDIM x K row-major, fp32) into WMMA B-fragment order (bf16 pairs).
// frag = ntg*(K/32)+kc ; word = lane*8+v ; n = ntg*16+(lane&15) ;
// k = kc*32 + (v>=4?16:0) + 2*(v&3) + 8*(lane>>4) ; word = {bf16(W[n][k+1]), bf16(W[n][k])}
// ---------------------------------------------------------------------------
__global__ void pack_weight_kernel(const float* __restrict__ W, uint32_t* __restrict__ dst, int K) {
  int idx = blockIdx.x * 256 + threadIdx.x;
  int total = GDIM * K / 2;
  if (idx >= total) return;
  int frag = idx >> 8;
  int rem  = idx & 255;
  int lane = rem >> 3;
  int v    = rem & 7;
  int KC   = K >> 5;
  int ntg  = frag / KC;
  int kc   = frag - ntg * KC;
  int n = ntg * 16 + (lane & 15);
  int k = kc * 32 + ((v >= 4) ? 16 : 0) + 2 * (v & 3) + 8 * (lane >> 4);
  uint32_t lo = f2bf(W[(size_t)n * K + k]);
  uint32_t hi = f2bf(W[(size_t)n * K + k + 1]);
  dst[idx] = lo | (hi << 16);
}

__global__ void bias_kernel(const float* b0a, const float* b0b, const float* b1a, const float* b1b,
                            float* bias0, float* bias1) {
  int i = blockIdx.x * 256 + threadIdx.x;
  if (i < GDIM) { bias0[i] = b0a[i] + b0b[i]; bias1[i] = b1a[i] + b1b[i]; }
}

// ---------------------------------------------------------------------------
// Feature pipeline -> feats bf16 in (T, BATCH, FDIM) layout.
// ---------------------------------------------------------------------------
__global__ void feat_kernel(const float* __restrict__ x_feat, const int* __restrict__ access_type,
                            const float* __restrict__ de_w1, const float* __restrict__ de_b1,
                            const float* __restrict__ de_w2, const float* __restrict__ de_b2,
                            const float* __restrict__ embed,
                            const float* __restrict__ comb_w, const float* __restrict__ comb_b,
                            uint16_t* __restrict__ feats) {
  __shared__ float s_w1[32 * 8], s_b1[32], s_w2[64 * 32], s_b2[64];
  __shared__ float s_emb[16 * 16], s_cw[64 * 80], s_cb[64];
  int tid = threadIdx.x;
  for (int i = tid; i < 256;  i += blockDim.x) s_w1[i]  = de_w1[i];
  for (int i = tid; i < 32;   i += blockDim.x) s_b1[i]  = de_b1[i];
  for (int i = tid; i < 2048; i += blockDim.x) s_w2[i]  = de_w2[i];
  for (int i = tid; i < 64;   i += blockDim.x) s_b2[i]  = de_b2[i];
  for (int i = tid; i < 256;  i += blockDim.x) s_emb[i] = embed[i];
  for (int i = tid; i < 5120; i += blockDim.x) s_cw[i]  = comb_w[i];
  for (int i = tid; i < 64;   i += blockDim.x) s_cb[i]  = comb_b[i];
  __syncthreads();

  int gid = blockIdx.x * blockDim.x + tid;
  if (gid >= BATCH * TSEQ) return;
  int b = gid / TSEQ, t = gid - b * TSEQ;

  const float* x = x_feat + (size_t)(b * TSEQ + t) * 8;
  float h1[32];
#pragma unroll
  for (int j = 0; j < 32; ++j) {
    float a = s_b1[j];
#pragma unroll
    for (int i = 0; i < 8; ++i) a += x[i] * s_w1[j * 8 + i];
    h1[j] = fmaxf(a, 0.0f);
  }
  float d[64];
#pragma unroll
  for (int j = 0; j < 64; ++j) {
    float a = s_b2[j];
#pragma unroll
    for (int i = 0; i < 32; ++i) a += h1[i] * s_w2[j * 32 + i];
    d[j] = a;
  }
  int at = access_type[b * TSEQ + t];
  const float* e = s_emb + at * 16;
  uint16_t* outp = feats + ((size_t)t * BATCH + b) * FDIM;
  for (int j = 0; j < 64; ++j) {
    const float* wr = s_cw + j * 80;
    float a = s_cb[j];
#pragma unroll
    for (int i = 0; i < 64; ++i) a += d[i] * wr[i];
#pragma unroll
    for (int i = 0; i < 16; ++i) a += e[i] * wr[64 + i];
    outp[j] = f2bf(a);
  }
}

// ---------------------------------------------------------------------------
// LSTM layer. 256 threads = 8 waves; BM=64 rows per WG. Packed weights in LDS.
// Wave owns rowtile (wv&3) and 4 H-ntiles; all 4 gate fragments per cell tile
// accumulate in registers (zero-init -> first WMMA uses inline-0 C, bias folded
// into the elementwise phase). x tiles double-buffered via async global->LDS.
// ---------------------------------------------------------------------------
template <int K_IN, bool HAS_YOUT, bool HAS_HFIN>
__global__ __launch_bounds__(256, 1) void lstm_layer_kernel(
    const uint16_t* __restrict__ xin,    // (TSEQ, BATCH, K_IN) bf16
    const uint32_t* __restrict__ wih_pk, // 32*(K_IN/32) frags * 256 words
    const uint32_t* __restrict__ whh_pk, // 32*4 frags * 256 words
    const float* __restrict__ bias,      // GDIM (bih+bhh)
    uint16_t* __restrict__ yout,         // (TSEQ, BATCH, HDIM) bf16
    float* __restrict__ hfinal)          // (BATCH, HDIM) fp32
{
  constexpr int KC_IN = K_IN / 32;
  constexpr int WIH_WORDS = 32 * KC_IN * 256;
  constexpr int WHH_WORDS = 32 * 4 * 256;
  constexpr int XBYTES = BM * K_IN * 2;
  extern __shared__ __align__(16) char smem[];
  uint32_t* wih_lds  = (uint32_t*)smem;
  uint32_t* whh_lds  = wih_lds + WIH_WORDS;
  float*    bias_lds = (float*)(whh_lds + WHH_WORDS);
  uint16_t* x_stage0 = (uint16_t*)(bias_lds + GDIM);  // BM x K_IN bf16 (buf 0)
  uint16_t* x_stage1 = x_stage0 + BM * K_IN;          // BM x K_IN bf16 (buf 1)
  uint16_t* h_stage  = x_stage1 + BM * K_IN;          // BM x HDIM bf16

  const int tid  = threadIdx.x;
  const int lane = tid & 31;
  const int wv   = tid >> 5;
  const int l15  = lane & 15;
  const int g2   = lane >> 4;
  const int b0   = blockIdx.x * BM;

  // prefetch first x tile through the async DMA path while weights load
  async_copy_tile(xin + (size_t)b0 * K_IN, x_stage0, XBYTES, tid);

  for (int i = tid; i < WIH_WORDS; i += 256) wih_lds[i] = wih_pk[i];
  for (int i = tid; i < WHH_WORDS; i += 256) whh_lds[i] = whh_pk[i];
  for (int i = tid; i < GDIM; i += 256) bias_lds[i] = bias[i];
  { uint32_t* hz = (uint32_t*)h_stage;
    for (int i = tid; i < BM * HDIM / 2; i += 256) hz[i] = 0u; }
  wait_async0();
  __syncthreads();

  const int rt  = wv & 3;          // rowtile 0..3
  const int ntb = (wv >> 2) * 4;   // first H-ntile (0 or 4)
  const int m0  = rt * 16;
  const int am  = m0 + l15;        // A-fragment row for this lane

  // hoist t-invariant bias scalars into registers
  float bvals[4][4];
#pragma unroll
  for (int ct = 0; ct < 4; ++ct) {
    const int n = (ntb + ct) * 16 + l15;
#pragma unroll
    for (int gi = 0; gi < 4; ++gi) bvals[ct][gi] = bias_lds[gi * HDIM + n];
  }

  float cfrag[4][8];
#pragma unroll
  for (int ct = 0; ct < 4; ++ct)
#pragma unroll
    for (int r = 0; r < 8; ++r) cfrag[ct][r] = 0.0f;

  for (int t = 0; t < TSEQ; ++t) {
    const uint16_t* xbuf = (t & 1) ? x_stage1 : x_stage0;
    uint16_t* xnext      = (t & 1) ? x_stage0 : x_stage1;
    if (t + 1 < TSEQ)
      async_copy_tile(xin + ((size_t)(t + 1) * BATCH + b0) * K_IN, xnext, XBYTES, tid);

    v8f acc[4][4];
#pragma unroll
    for (int ct = 0; ct < 4; ++ct)
#pragma unroll
      for (int gi = 0; gi < 4; ++gi) acc[ct][gi] = (v8f){0,0,0,0,0,0,0,0};

    // input GEMM: x_t @ wih.T
#pragma unroll
    for (int kc = 0; kc < KC_IN; ++kc) {
      v16bf af = load_A_frag(xbuf, K_IN, am, kc * 32, g2);
#pragma unroll
      for (int ct = 0; ct < 4; ++ct)
#pragma unroll
        for (int gi = 0; gi < 4; ++gi) {
          v16bf bf = load_B_frag(wih_lds, (gi * 8 + ntb + ct) * KC_IN + kc, lane);
          acc[ct][gi] = wmma_bf16(af, bf, acc[ct][gi]);
        }
    }
    // recurrent GEMM: h_{t-1} @ whh.T
#pragma unroll
    for (int kc = 0; kc < 4; ++kc) {
      v16bf af = load_A_frag(h_stage, HDIM, am, kc * 32, g2);
#pragma unroll
      for (int ct = 0; ct < 4; ++ct)
#pragma unroll
        for (int gi = 0; gi < 4; ++gi) {
          v16bf bf = load_B_frag(whh_lds, (gi * 8 + ntb + ct) * 4 + kc, lane);
          acc[ct][gi] = wmma_bf16(af, bf, acc[ct][gi]);
        }
    }
    __syncthreads();   // all h_stage / xbuf reads done

    // elementwise LSTM cell (bias folded in here; four gate frags share (M,N) map)
#pragma unroll
    for (int ct = 0; ct < 4; ++ct) {
      const int n = (ntb + ct) * 16 + l15;
#pragma unroll
      for (int r = 0; r < 8; ++r) {
        const float ig = sigm(acc[ct][0][r] + bvals[ct][0]);
        const float fg = sigm(acc[ct][1][r] + bvals[ct][1]);
        const float gg = tanhf(acc[ct][2][r] + bvals[ct][2]);
        const float og = sigm(acc[ct][3][r] + bvals[ct][3]);
        const float cn = fg * cfrag[ct][r] + ig * gg;
        cfrag[ct][r] = cn;
        const float hn = og * tanhf(cn);
        const int m = m0 + r + 8 * g2;          // C-layout: VGPR r -> M=r (+8 hi lanes)
        const uint16_t hb = f2bf(hn);
        h_stage[m * HDIM + n] = hb;
        if (HAS_YOUT) yout[((size_t)t * BATCH + b0 + m) * HDIM + n] = hb;
        if (HAS_HFIN && t == TSEQ - 1) hfinal[(size_t)(b0 + m) * HDIM + n] = hn;
      }
    }
    wait_async0();     // next x tile landed (this wave's async ops)
    __syncthreads();   // h_stage + next x tile visible WG-wide
  }
}

// ---------------------------------------------------------------------------
// Heads: deltas(128->128->8)*1e6 ; conf sigmoid(128->64->8) ; probs softmax(128->32->4)
// ---------------------------------------------------------------------------
__global__ void head_kernel(const float* __restrict__ hfinal,
                            const float* __restrict__ dp_w1, const float* __restrict__ dp_b1,
                            const float* __restrict__ dp_w2, const float* __restrict__ dp_b2,
                            const float* __restrict__ cp_w1, const float* __restrict__ cp_b1,
                            const float* __restrict__ cp_w2, const float* __restrict__ cp_b2,
                            const float* __restrict__ pc_w1, const float* __restrict__ pc_b1,
                            const float* __restrict__ pc_w2, const float* __restrict__ pc_b2,
                            float* __restrict__ out) {
  __shared__ float sh[128];
  __shared__ float hid[128];
  __shared__ float outs[20];
  const int b = blockIdx.x;
  const int tid = threadIdx.x;
  sh[tid] = hfinal[(size_t)b * HDIM + tid];
  __syncthreads();
  { float a = dp_b1[tid];
    for (int i = 0; i < 128; ++i) a += sh[i] * dp_w1[tid * 128 + i];
    hid[tid] = fmaxf(a, 0.0f); }
  __syncthreads();
  if (tid < 8) {
    float s = dp_b2[tid];
    for (int i = 0; i < 128; ++i) s += hid[i] * dp_w2[tid * 128 + i];
    outs[tid] = s * 1000000.0f;
  }
  __syncthreads();
  if (tid < 64) {
    float a = cp_b1[tid];
    for (int i = 0; i < 128; ++i) a += sh[i] * cp_w1[tid * 128 + i];
    hid[tid] = fmaxf(a, 0.0f);
  }
  __syncthreads();
  if (tid < 8) {
    float s = cp_b2[tid];
    for (int i = 0; i < 64; ++i) s += hid[i] * cp_w2[tid * 64 + i];
    outs[8 + tid] = sigm(s);
  }
  __syncthreads();
  if (tid < 32) {
    float a = pc_b1[tid];
    for (int i = 0; i < 128; ++i) a += sh[i] * pc_w1[tid * 128 + i];
    hid[tid] = fmaxf(a, 0.0f);
  }
  __syncthreads();
  if (tid < 4) {
    float s = pc_b2[tid];
    for (int i = 0; i < 32; ++i) s += hid[i] * pc_w2[tid * 32 + i];
    outs[16 + tid] = s;
  }
  __syncthreads();
  if (tid == 0) {
    float m = outs[16];
    for (int i = 17; i < 20; ++i) m = fmaxf(m, outs[i]);
    float sum = 0.0f;
    for (int i = 16; i < 20; ++i) { outs[i] = __expf(outs[i] - m); sum += outs[i]; }
    float inv = 1.0f / sum;
    for (int i = 16; i < 20; ++i) outs[i] *= inv;
  }
  __syncthreads();
  if (tid < 20) out[(size_t)b * 20 + tid] = outs[tid];
}

// ---------------------------------------------------------------------------
extern "C" void kernel_launch(void* const* d_in, const int* in_sizes, int n_in,
                              void* d_out, int out_size, void* d_ws, size_t ws_size,
                              hipStream_t stream) {
  (void)in_sizes; (void)n_in; (void)out_size; (void)ws_size;
  const float* x_feat      = (const float*)d_in[0];
  const int*   access_type = (const int*)  d_in[1];
  const float* de_w1 = (const float*)d_in[2];
  const float* de_b1 = (const float*)d_in[3];
  const float* de_w2 = (const float*)d_in[4];
  const float* de_b2 = (const float*)d_in[5];
  const float* embed = (const float*)d_in[6];
  const float* comb_w = (const float*)d_in[7];
  const float* comb_b = (const float*)d_in[8];
  const float* lstm0_wih = (const float*)d_in[9];
  const float* lstm0_whh = (const float*)d_in[10];
  const float* lstm0_bih = (const float*)d_in[11];
  const float* lstm0_bhh = (const float*)d_in[12];
  const float* lstm1_wih = (const float*)d_in[13];
  const float* lstm1_whh = (const float*)d_in[14];
  const float* lstm1_bih = (const float*)d_in[15];
  const float* lstm1_bhh = (const float*)d_in[16];
  const float* dp_w1 = (const float*)d_in[17];
  const float* dp_b1 = (const float*)d_in[18];
  const float* dp_w2 = (const float*)d_in[19];
  const float* dp_b2 = (const float*)d_in[20];
  const float* cp_w1 = (const float*)d_in[21];
  const float* cp_b1 = (const float*)d_in[22];
  const float* cp_w2 = (const float*)d_in[23];
  const float* cp_b2 = (const float*)d_in[24];
  const float* pc_w1 = (const float*)d_in[25];
  const float* pc_b1 = (const float*)d_in[26];
  const float* pc_w2 = (const float*)d_in[27];
  const float* pc_b2 = (const float*)d_in[28];

  char* ws = (char*)d_ws;
  size_t off = 0;
  uint16_t* feats = (uint16_t*)(ws + off); off += (size_t)TSEQ * BATCH * FDIM * 2;   // 32 MB
  uint16_t* y0    = (uint16_t*)(ws + off); off += (size_t)TSEQ * BATCH * HDIM * 2;   // 64 MB
  float*    hfin  = (float*)   (ws + off); off += (size_t)BATCH * HDIM * 4;          // 4 MB
  uint32_t* wih0_pk = (uint32_t*)(ws + off); off += (size_t)GDIM * 64 * 2;   // 64 KB
  uint32_t* whh0_pk = (uint32_t*)(ws + off); off += (size_t)GDIM * 128 * 2;  // 128 KB
  uint32_t* wih1_pk = (uint32_t*)(ws + off); off += (size_t)GDIM * 128 * 2;
  uint32_t* whh1_pk = (uint32_t*)(ws + off); off += (size_t)GDIM * 128 * 2;
  float* bias0 = (float*)(ws + off); off += GDIM * 4;
  float* bias1 = (float*)(ws + off); off += GDIM * 4;

  // 1) weight prep
  pack_weight_kernel<<<(GDIM * 64 / 2 + 255) / 256, 256, 0, stream>>>(lstm0_wih, wih0_pk, 64);
  pack_weight_kernel<<<(GDIM * 128 / 2 + 255) / 256, 256, 0, stream>>>(lstm0_whh, whh0_pk, 128);
  pack_weight_kernel<<<(GDIM * 128 / 2 + 255) / 256, 256, 0, stream>>>(lstm1_wih, wih1_pk, 128);
  pack_weight_kernel<<<(GDIM * 128 / 2 + 255) / 256, 256, 0, stream>>>(lstm1_whh, whh1_pk, 128);
  bias_kernel<<<2, 256, 0, stream>>>(lstm0_bih, lstm0_bhh, lstm1_bih, lstm1_bhh, bias0, bias1);

  // 2) feature MLP -> feats (T,B,F) bf16
  feat_kernel<<<(BATCH * TSEQ + 255) / 256, 256, 0, stream>>>(
      x_feat, access_type, de_w1, de_b1, de_w2, de_b2, embed, comb_w, comb_b, feats);

  // 3) LSTM layers (WMMA bf16, weights in LDS, async double-buffered x tiles)
  const size_t smem0 = (size_t)GDIM * 64 * 2 + (size_t)GDIM * 128 * 2 + GDIM * 4
                     + 2 * (size_t)BM * 64 * 2 + (size_t)BM * HDIM * 2;   // 231424 B
  const size_t smem1 = (size_t)GDIM * 128 * 2 + (size_t)GDIM * 128 * 2 + GDIM * 4
                     + 2 * (size_t)BM * 128 * 2 + (size_t)BM * HDIM * 2;  // 313344 B
  (void)hipFuncSetAttribute((const void*)&lstm_layer_kernel<64, true, false>,
                            hipFuncAttributeMaxDynamicSharedMemorySize, (int)smem0);
  (void)hipFuncSetAttribute((const void*)&lstm_layer_kernel<128, false, true>,
                            hipFuncAttributeMaxDynamicSharedMemorySize, (int)smem1);

  lstm_layer_kernel<64, true, false><<<BATCH / BM, 256, smem0, stream>>>(
      feats, wih0_pk, whh0_pk, bias0, y0, (float*)nullptr);
  lstm_layer_kernel<128, false, true><<<BATCH / BM, 256, smem1, stream>>>(
      y0, wih1_pk, whh1_pk, bias1, (uint16_t*)nullptr, hfin);

  // 4) heads
  head_kernel<<<BATCH, 128, 0, stream>>>(hfin, dp_w1, dp_b1, dp_w2, dp_b2,
                                         cp_w1, cp_b1, cp_w2, cp_b2,
                                         pc_w1, pc_b1, pc_w2, pc_b2, (float*)d_out);
}
